// LinearAttention_24223615550350
// MI455X (gfx1250) — matrix operand order repaired and checked by reading, not compiled, not run on hardware
//
#include <hip/hip_runtime.h>

#define SEQ    4096
#define DIM    64
#define BH     64        // B*H = 4*16
#define NTHR   512       // 16 wave32
#define NWAVE  16
#define CHUNK  32        // key rows staged per phase-3 step
#define NCHUNK (SEQ / CHUNK)

typedef __attribute__((ext_vector_type(16))) __bf16 v16bf;
typedef __attribute__((ext_vector_type(8)))  float  v8f;

struct SMem {
  float p[SEQ];                   // 16 KB: scores -> softmax weights
  float stageK[2][CHUNK][DIM];    // 16 KB: raw phi_K chunks (async-DMA dest)
  float stageV[2][CHUNK][DIM];    // 16 KB: raw V chunks (async-DMA dest)
  float red[NTHR];                // 2 KB : block reductions
  float qprobe[DIM];
  float KV[DIM][DIM];             // 16 KB: accumulated (p.phiK)^T V
  float ksum[DIM];
};                                // ~68 KB of the 320 KB WGP LDS

__global__ __launch_bounds__(NTHR, 1) void
linattn_fused(const float* __restrict__ Q,  const float* __restrict__ Kmat,
              const float* __restrict__ V,  const float* __restrict__ phiQ,
              const float* __restrict__ phiK, float* __restrict__ out)
{
  __shared__ SMem sm;
  const int tid   = threadIdx.x;
  const int lane  = tid & 31;
  const int wave  = tid >> 5;
  const int m16   = lane & 15;   // N (or M) index inside a 16x16 tile
  const int khalf = lane >> 4;   // lane-group selector in wave32 WMMA layouts
  const size_t base = (size_t)blockIdx.x * SEQ * DIM;

  const float* Qp = Q    + base;
  const float* Kp = Kmat + base;
  const float* Vp = V    + base;
  const float* pQ = phiQ + base;
  const float* pK = phiK + base;
  float*       Op = out  + base;

  // one b128 async transfer per thread per array per chunk
  const int rowA = tid >> 4;          // 0..31 : staged row
  const int col4 = (tid & 15) * 4;    // float column of this thread's 16B

  auto issue_async = [&](int buf, int c) {
    const size_t g = (size_t)(c * CHUNK + rowA) * DIM + col4;
    const unsigned ldsK = (unsigned)(unsigned long long)&sm.stageK[buf][rowA][col4];
    const unsigned ldsV = (unsigned)(unsigned long long)&sm.stageV[buf][rowA][col4];
    asm volatile("global_load_async_to_lds_b128 %0, %1, off"
                 :: "v"(ldsK), "v"(pK + g) : "memory");
    asm volatile("global_load_async_to_lds_b128 %0, %1, off"
                 :: "v"(ldsV), "v"(Vp + g) : "memory");
  };

  //------------------------------------------------------------------
  // Phase 1: q_probe = mean_s(Q) / sqrt(D)        (reads Q once)
  //------------------------------------------------------------------
  {
    const int col  = tid & (DIM - 1);
    const int rgrp = tid >> 6;                  // 0..7
    float acc = 0.f;
    for (int s = rgrp; s < SEQ; s += 8)
      acc += Qp[(size_t)s * DIM + col];
    sm.red[tid] = acc;
    __syncthreads();
    if (tid < DIM) {
      float v = 0.f;
      #pragma unroll
      for (int j = 0; j < 8; ++j) v += sm.red[tid + j * DIM];
      sm.qprobe[tid] = v * (1.0f / (SEQ * 8.0f)); // /S and /sqrt(64)
    }
    __syncthreads();
  }

  // kick off the first phase-3 chunk now: the DMA hides under phase 2
  issue_async(0, 0);

  //------------------------------------------------------------------
  // Phase 2: p = softmax_s(K @ q_probe)           (reads K once)
  //------------------------------------------------------------------
  {
    const float q0 = sm.qprobe[lane];
    const float q1 = sm.qprobe[lane + 32];
    for (int s = wave; s < SEQ; s += NWAVE) {
      const float* kr = Kp + (size_t)s * DIM;
      float x = kr[lane] * q0 + kr[lane + 32] * q1;
      #pragma unroll
      for (int off = 16; off > 0; off >>= 1)
        x += __shfl_xor(x, off, 32);
      if (lane == 0) sm.p[s] = x;
    }
    __syncthreads();
    float m = -3.4e38f;
    for (int s = tid; s < SEQ; s += NTHR) m = fmaxf(m, sm.p[s]);
    sm.red[tid] = m;
    __syncthreads();
    for (int w = NTHR / 2; w >= 1; w >>= 1) {
      if (tid < w) sm.red[tid] = fmaxf(sm.red[tid], sm.red[tid + w]);
      __syncthreads();
    }
    const float gmax = sm.red[0];
    __syncthreads();
    float es = 0.f;
    for (int s = tid; s < SEQ; s += NTHR) {
      const float e = __expf(sm.p[s] - gmax);
      sm.p[s] = e;
      es += e;
    }
    sm.red[tid] = es;
    __syncthreads();
    for (int w = NTHR / 2; w >= 1; w >>= 1) {
      if (tid < w) sm.red[tid] += sm.red[tid + w];
      __syncthreads();
    }
    const float inv = 1.0f / sm.red[0];
    for (int s = tid; s < SEQ; s += NTHR) sm.p[s] *= inv;
    __syncthreads();
  }

  //------------------------------------------------------------------
  // Phase 3: KV = (p.phiK)^T V  via WMMA, async double-buffered LDS
  //          staging; ksum scalar-exact   (reads phi_K and V once)
  //------------------------------------------------------------------
  v8f acc = {};
  float ks = 0.f;
  const int ti = wave >> 2;      // tile row of this wave's 16x16 KV tile
  const int tj = wave & 3;       // tile col
  {
    const int col  = tid & (DIM - 1);
    const int rgrp = tid >> 6;   // 0..7
    for (int c = 0; c < NCHUNK; ++c) {
      const int buf = c & 1;
      const int s0  = c * CHUNK;
      if (c + 1 < NCHUNK) {
        issue_async(buf ^ 1, c + 1);              // prefetch next chunk
        asm volatile("s_wait_asynccnt 0x2" ::: "memory"); // this chunk landed
      } else {
        asm volatile("s_wait_asynccnt 0x0" ::: "memory");
      }
      __syncthreads();                            // publish DMA'd LDS data

      // ksum partials straight from staged phi_K
      #pragma unroll
      for (int j = 0; j < 4; ++j) {
        const int r = rgrp + j * 8;               // 0..31
        ks += sm.stageK[buf][r][col] * sm.p[s0 + r];
      }

      // A[m][k] = p[s0+k] * phiK[s0+k][16*ti + m]  (transposed, scaled)
      v16bf afrag, bfrag;
      #pragma unroll
      for (int e = 0; e < 8; ++e) {
        const int k0 = (e < 4 ? 2 * e : 16 + 2 * (e - 4)) + khalf * 8;
        afrag[2 * e]     = (__bf16)(sm.stageK[buf][k0    ][ti * 16 + m16] * sm.p[s0 + k0    ]);
        afrag[2 * e + 1] = (__bf16)(sm.stageK[buf][k0 + 1][ti * 16 + m16] * sm.p[s0 + k0 + 1]);
      }
      // B[k][n] = V[s0+k][16*tj + n]
      #pragma unroll
      for (int e = 0; e < 16; ++e)
        bfrag[e] = (__bf16)sm.stageV[buf][khalf * 8 + e + (e < 8 ? 0 : 8)][tj * 16 + m16];

      acc = __builtin_amdgcn_wmma_f32_16x16x32_bf16(
              false, afrag, false, bfrag, (short)0, acc, false, false);
      __syncthreads();                            // readers done before re-DMA
    }
  }
  // fold ksum partials; park KV tile (fp32 accumulators) in LDS
  sm.red[tid] = ks;
  __syncthreads();
  if (tid < DIM) {
    float v = 0.f;
    #pragma unroll
    for (int j = 0; j < 8; ++j) v += sm.red[tid + j * DIM];
    sm.ksum[tid] = v;
  }
  #pragma unroll
  for (int r = 0; r < 8; ++r)
    sm.KV[ti * 16 + r + 8 * khalf][tj * 16 + m16] = acc[r];
  __syncthreads();

  //------------------------------------------------------------------
  // Phase 4: out = (phiQ @ KV) / (phiQ . ksum + 1e-6)
  //          (reads phi_Q once, writes out once)
  //------------------------------------------------------------------
  // KV -> 8 resident bf16 B-fragments per wave (reused for all strips)
  v16bf bKV[2][4];
  #pragma unroll
  for (int kc = 0; kc < 2; ++kc)
    #pragma unroll
    for (int n = 0; n < 4; ++n)
      #pragma unroll
      for (int e = 0; e < 16; ++e)
        bKV[kc][n][e] = (__bf16)sm.KV[kc * 32 + khalf * 8 + e + (e < 8 ? 0 : 8)][n * 16 + m16];

  for (int it = 0; it < SEQ / (NWAVE * 16); ++it) {
    const int sb = (it * NWAVE + wave) * 16;
    const float* q = pQ + (size_t)(sb + m16) * DIM;
    if (it + 1 < SEQ / (NWAVE * 16))              // prefetch next strip row
      __builtin_prefetch(q + (size_t)NWAVE * 16 * DIM, 0, 1);
    // exact fp32 denominator; lane L holds row sb + (L&15)
    float dn = 0.f;
    #pragma unroll 8
    for (int d = 0; d < DIM; ++d) dn += q[d] * sm.ksum[d];
    const float dinv = 1.0f / (dn + 1e-6f);
    // A fragments: phiQ strip, K-chunks 0..31 and 32..63
    v16bf a0, a1;
    #pragma unroll
    for (int e = 0; e < 8; ++e) {
      const int k0 = (e < 4 ? 2 * e : 16 + 2 * (e - 4)) + khalf * 8;
      a0[2 * e]     = (__bf16)q[k0];
      a0[2 * e + 1] = (__bf16)q[k0 + 1];
      a1[2 * e]     = (__bf16)q[32 + k0];
      a1[2 * e + 1] = (__bf16)q[32 + k0 + 1];
    }
    #pragma unroll
    for (int n = 0; n < 4; ++n) {
      v8f c = {};
      c = __builtin_amdgcn_wmma_f32_16x16x32_bf16(
            false, a0, false, bKV[0][n], (short)0, c, false, false);
      c = __builtin_amdgcn_wmma_f32_16x16x32_bf16(
            false, a1, false, bKV[1][n], (short)0, c, false, false);
      #pragma unroll
      for (int r = 0; r < 8; ++r) {
        const int M = r + 8 * khalf;
        const float d = __shfl(dinv, M, 32);   // row M's denominator
        Op[(size_t)(sb + M) * DIM + n * 16 + m16] = c[r] * d;
      }
    }
  }
}

extern "C" void kernel_launch(void* const* d_in, const int* in_sizes, int n_in,
                              void* d_out, int out_size, void* d_ws, size_t ws_size,
                              hipStream_t stream) {
  const float* Q   = (const float*)d_in[0];
  const float* K   = (const float*)d_in[1];
  const float* V   = (const float*)d_in[2];
  const float* pQ  = (const float*)d_in[3];
  const float* pK  = (const float*)d_in[4];
  float* out = (float*)d_out;
  (void)in_sizes; (void)n_in; (void)out_size; (void)d_ws; (void)ws_size;
  linattn_fused<<<dim3(BH), dim3(NTHR), 0, stream>>>(Q, K, V, pQ, pK, out);
}
